// MultiHeadAttention_13683765805642
// MI455X (gfx1250) — compile-verified, matrix operand order checked by reading, not run on hardware
//
#include <hip/hip_runtime.h>
#include <hip/hip_bf16.h>
#include <math.h>

typedef __attribute__((ext_vector_type(16))) _Float16 v16h;
typedef __attribute__((ext_vector_type(8)))  float    v8f;

#define B_ 4
#define S_ 2048
#define DM_ 512
#define H_ 8
#define DEPTH_ 64
#define LOG2E_ 1.4426950408889634f

__device__ __forceinline__ v8f wmma_f16(v16h a, v16h b, v8f c) {
  // D = A(16x32 f16) * B(32x16 f16) + C(16x16 f32)
  return __builtin_amdgcn_wmma_f32_16x16x32_f16(false, a, false, b, (short)0, c,
                                                false, false);
}

// A-fragment loader (16x32 f16, row-major source, row stride = ld elements).
// Lane L: row m = L&15, K-base kb = (L>>4)*8; VGPR0..3 <- K=kb..kb+7,
// VGPR4..7 <- K=16+kb..16+kb+7  (two contiguous 16B chunks -> b128 loads).
// The B-fragment (32x16, K x N) of matrix M equals the A-fragment of M^T,
// so transposed operands (K tile, Wt, Vt) use this same loader.
__device__ __forceinline__ v16h load_frag_a(const _Float16* __restrict__ p,
                                            int ld, int lane) {
  const int m  = lane & 15;
  const int kb = (lane >> 4) << 3;
  const _Float16* r = p + m * ld + kb;
  v16h f;
#pragma unroll
  for (int i = 0; i < 8; ++i) f[i] = r[i];
#pragma unroll
  for (int i = 0; i < 8; ++i) f[8 + i] = r[16 + i];
  return f;
}

// ---------------- conversion kernels ----------------
__global__ void cvt_f32_to_f16(const float* __restrict__ in,
                               _Float16* __restrict__ out, int n) {
  int i = blockIdx.x * blockDim.x + threadIdx.x;
  int stride = gridDim.x * blockDim.x;
  for (; i < n; i += stride)
    out[i] = (_Float16)__builtin_nontemporal_load(in + i);
}

// W [512x512] row-major -> Wt f16 with Wt[n*512+k] = W[k*512+n]
__global__ void cvt_transpose_512(const float* __restrict__ W,
                                  _Float16* __restrict__ Wt) {
  int i = blockIdx.x * blockDim.x + threadIdx.x;  // over 512*512
  if (i < DM_ * DM_) {
    int k = i >> 9, n = i & 511;
    Wt[n * DM_ + k] = (_Float16)W[i];
  }
}

// ---------------- GEMM: Y = X(8192x512,f16) @ W(512x512) + bias ----------------
// Wt is W transposed (f16).  MODE 0: f16 out, head-split [B,H,S,64] (Q/K)
//                            MODE 2: f16 out, transposed  [B,H,64,S] (V)
//                            MODE 3: f32 out, row-major [8192,512]   (output proj)
template <int MODE>
__global__ void gemm512_wmma(const _Float16* __restrict__ X,
                             const _Float16* __restrict__ Wt,
                             const float* __restrict__ bias,
                             _Float16* __restrict__ out16,
                             float* __restrict__ out32) {
  const int lane = threadIdx.x & 31;
  const int wave = threadIdx.x >> 5;
  const int M0 = blockIdx.x * 16;   // 512 row-tiles
  const int N0 = wave * 64;         // 8 waves cover 512 cols

  v8f acc[4] = {v8f{}, v8f{}, v8f{}, v8f{}};
#pragma unroll 4
  for (int kk = 0; kk < DM_; kk += 32) {
    v16h a = load_frag_a(X + (size_t)M0 * DM_ + kk, DM_, lane);
#pragma unroll
    for (int t = 0; t < 4; ++t) {
      v16h b = load_frag_a(Wt + (size_t)(N0 + t * 16) * DM_ + kk, DM_, lane);
      acc[t] = wmma_f16(a, b, acc[t]);
    }
  }

  const int nloc = lane & 15;
  const int mloc = (lane >> 4) << 3;
#pragma unroll
  for (int t = 0; t < 4; ++t) {
    const int n = N0 + t * 16 + nloc;
    const float bv = bias[n];
#pragma unroll
    for (int r = 0; r < 8; ++r) {
      const int m = M0 + mloc + r;           // global row in [0, 8192)
      const float val = acc[t][r] + bv;
      if (MODE == 3) {
        // final output: write-only, stream past L2
        __builtin_nontemporal_store(val, &out32[(size_t)m * DM_ + n]);
      } else {
        const int b = m >> 11, s = m & (S_ - 1);
        const int h = n >> 6,  d = n & (DEPTH_ - 1);
        if (MODE == 0)  // [B,H,S,64]
          out16[(((size_t)(b * H_ + h)) * S_ + s) * DEPTH_ + d] = (_Float16)val;
        else            // MODE 2: [B,H,64,S]
          out16[(((size_t)(b * H_ + h)) * DEPTH_ + d) * S_ + s] = (_Float16)val;
      }
    }
  }
}

// ---------------- attention: logits -> softmax -> attn out -> ctx ----------------
// One workgroup (8 wave32) per (b, h, 16-query tile).  Logits tile kept in LDS.
// Logits are stored pre-scaled by log2(e) so softmax uses v_exp_f32 (base-2)
// directly; normalization makes the base change exact.
__global__ void attn_wmma(const _Float16* __restrict__ Qh,
                          const _Float16* __restrict__ Kh,
                          const _Float16* __restrict__ Vt,
                          const float* __restrict__ mask,
                          float* __restrict__ attn_out,
                          _Float16* __restrict__ ctxh) {
  __shared__ float lds_l[16 * S_];       // 128 KB logits/prob tile
  __shared__ float lds_c[16 * DEPTH_];   // 4 KB ctx accumulator

  const int lane = threadIdx.x & 31;
  const int wave = threadIdx.x >> 5;
  const int bh = blockIdx.x >> 7;        // b*H + h
  const int qt = blockIdx.x & 127;       // query tile
  const int bidx = bh >> 3;
  const int h = bh & 7;

  const int nloc = lane & 15;
  const int mloc = (lane >> 4) << 3;

  // Q fragments for this 16-row tile (depth 0..31 and 32..63)
  const _Float16* Qbase = Qh + ((size_t)bh * S_ + qt * 16) * DEPTH_;
  const v16h qa0 = load_frag_a(Qbase, DEPTH_, lane);
  const v16h qa1 = load_frag_a(Qbase + 32, DEPTH_, lane);

  // ---- phase 1: logits2 = (Q K^T) * (log2e/8) + mask*(-1e9*log2e) into LDS ----
  for (int kt = wave; kt < S_ / 16; kt += 8) {
    const _Float16* Kbase = Kh + ((size_t)bh * S_ + kt * 16) * DEPTH_;
    // hint the next K tile for this wave (8 tiles = 16 KB ahead)
    if (kt + 8 < S_ / 16)
      __builtin_prefetch(Kbase + 8 * 16 * DEPTH_ + (lane << 6), 0, 1);
    v16h ka0 = load_frag_a(Kbase, DEPTH_, lane);
    v16h ka1 = load_frag_a(Kbase + 32, DEPTH_, lane);
    v8f c = v8f{};
    c = wmma_f16(qa0, ka0, c);
    c = wmma_f16(qa1, ka1, c);
    const int ncol = kt * 16 + nloc;
    const float mval = mask[bidx * S_ + ncol] * (-1e9f * LOG2E_);
#pragma unroll
    for (int r = 0; r < 8; ++r)
      lds_l[(mloc + r) * S_ + ncol] = c[r] * (0.125f * LOG2E_) + mval;
  }
  __syncthreads();

  // ---- phase 2: row softmax (base-2) + nontemporal coalesced attn store ----
#pragma unroll
  for (int rr = 0; rr < 2; ++rr) {
    const int row = wave * 2 + rr;
    float* L = lds_l + row * S_;
    float mx = -INFINITY;
    for (int j = lane; j < S_; j += 32) mx = fmaxf(mx, L[j]);
#pragma unroll
    for (int off = 16; off >= 1; off >>= 1)
      mx = fmaxf(mx, __shfl_xor(mx, off, 32));
    float sum = 0.f;
    for (int j = lane; j < S_; j += 32) {
      float e = exp2f(L[j] - mx);     // v_exp_f32 directly
      L[j] = e;
      sum += e;
    }
#pragma unroll
    for (int off = 16; off >= 1; off >>= 1) sum += __shfl_xor(sum, off, 32);
    const float inv = 1.0f / sum;
    float* A = attn_out + ((size_t)bh * S_ + qt * 16 + row) * S_;
    for (int j = lane; j < S_; j += 32) {
      float p = L[j] * inv;
      L[j] = p;
      __builtin_nontemporal_store(p, A + j);   // 537 MB stream: keep L2 for K/V
    }
  }

  // zero ctx accumulator
  for (int i = threadIdx.x; i < 16 * DEPTH_; i += 256) lds_c[i] = 0.f;
  __syncthreads();

  // ---- phase 3: ctx = P @ V, waves stripe K, reduce via ds_add_f32 ----
  v8f acc[4] = {v8f{}, v8f{}, v8f{}, v8f{}};
  for (int kt = wave; kt < S_ / 32; kt += 8) {
    // A fragment from LDS probabilities (f32 -> f16)
    const float* Lr = lds_l + nloc * S_ + kt * 32 + mloc; // row=lane&15, kb=(lane>>4)*8
    v16h a;
#pragma unroll
    for (int i = 0; i < 8; ++i) a[i] = (_Float16)Lr[i];
#pragma unroll
    for (int i = 0; i < 8; ++i) a[8 + i] = (_Float16)Lr[16 + i];
#pragma unroll
    for (int t = 0; t < 4; ++t) {
      const _Float16* Vb = Vt + ((size_t)bh * DEPTH_ + t * 16) * S_ + kt * 32;
      v16h b = load_frag_a(Vb, S_, lane);
      acc[t] = wmma_f16(a, b, acc[t]);
    }
  }
#pragma unroll
  for (int t = 0; t < 4; ++t)
#pragma unroll
    for (int r = 0; r < 8; ++r)
      atomicAdd(&lds_c[(mloc + r) * DEPTH_ + t * 16 + nloc], acc[t][r]);
  __syncthreads();

  // ---- phase 4: ctx -> f16, merged-head layout [B,S,512] ----
  for (int i = threadIdx.x; i < 16 * DEPTH_; i += 256) {
    const int row = i >> 6, d = i & 63;
    ctxh[((size_t)(bidx * S_ + qt * 16 + row)) * DM_ + h * DEPTH_ + d] =
        (_Float16)lds_c[i];
  }
}

// ---------------- host side ----------------
extern "C" void kernel_launch(void* const* d_in, const int* in_sizes, int n_in,
                              void* d_out, int out_size, void* d_ws, size_t ws_size,
                              hipStream_t stream) {
  const float* q    = (const float*)d_in[0];
  const float* k    = (const float*)d_in[1];
  const float* v    = (const float*)d_in[2];
  const float* mask = (const float*)d_in[3];
  const float* Wq = (const float*)d_in[4];   const float* bq = (const float*)d_in[5];
  const float* Wk = (const float*)d_in[6];   const float* bk = (const float*)d_in[7];
  const float* Wv = (const float*)d_in[8];   const float* bv = (const float*)d_in[9];
  const float* Wo = (const float*)d_in[10];  const float* bo = (const float*)d_in[11];

  float* out  = (float*)d_out;                       // [4,2048,512]
  float* attn = out + (size_t)B_ * S_ * DM_;         // [4,8,2048,2048]

  // workspace carve-up (f16 buffers)
  const size_t NX = (size_t)B_ * S_ * DM_;           // 4,194,304
  char* ws = (char*)d_ws;
  _Float16* qf  = (_Float16*)ws;            ws += NX * 2;
  _Float16* kf  = (_Float16*)ws;            ws += NX * 2;
  _Float16* vf  = (_Float16*)ws;            ws += NX * 2;
  _Float16* WqT = (_Float16*)ws;            ws += (size_t)DM_ * DM_ * 2;
  _Float16* WkT = (_Float16*)ws;            ws += (size_t)DM_ * DM_ * 2;
  _Float16* WvT = (_Float16*)ws;            ws += (size_t)DM_ * DM_ * 2;
  _Float16* WoT = (_Float16*)ws;            ws += (size_t)DM_ * DM_ * 2;
  _Float16* Qh  = (_Float16*)ws;            ws += NX * 2;
  _Float16* Kh  = (_Float16*)ws;            ws += NX * 2;
  _Float16* Vt  = (_Float16*)ws;            ws += NX * 2;
  _Float16* ctx = (_Float16*)ws;            ws += NX * 2;

  const int nconv = (int)NX;
  cvt_f32_to_f16<<<2048, 256, 0, stream>>>(q, qf, nconv);
  cvt_f32_to_f16<<<2048, 256, 0, stream>>>(k, kf, nconv);
  cvt_f32_to_f16<<<2048, 256, 0, stream>>>(v, vf, nconv);

  const int nw = DM_ * DM_;
  cvt_transpose_512<<<(nw + 255) / 256, 256, 0, stream>>>(Wq, WqT);
  cvt_transpose_512<<<(nw + 255) / 256, 256, 0, stream>>>(Wk, WkT);
  cvt_transpose_512<<<(nw + 255) / 256, 256, 0, stream>>>(Wv, WvT);
  cvt_transpose_512<<<(nw + 255) / 256, 256, 0, stream>>>(Wo, WoT);

  // projections: 512 row-tiles of 16, 8 waves cover 512 cols
  gemm512_wmma<0><<<512, 256, 0, stream>>>(qf, WqT, bq, Qh, nullptr);
  gemm512_wmma<0><<<512, 256, 0, stream>>>(kf, WkT, bk, Kh, nullptr);
  gemm512_wmma<2><<<512, 256, 0, stream>>>(vf, WvT, bv, Vt, nullptr);

  // attention: one block per (b,h,16-query tile) = 4*8*128 = 4096 blocks
  attn_wmma<<<B_ * H_ * (S_ / 16), 256, 0, stream>>>(Qh, Kh, Vt, mask, attn, ctx);

  // output projection (f32 out)
  gemm512_wmma<3><<<512, 256, 0, stream>>>(ctx, WoT, bo, nullptr, out);
}
